// BinaryXnorExceptOutliersLinearColumn_30416958390730
// MI455X (gfx1250) — compile-verified, matrix operand order checked by reading, not run on hardware
//
#include <hip/hip_runtime.h>
#include <hip/hip_bf16.h>

// Shapes (fixed by the reference)
#define M_ROWS 256      // B*S = 32*8
#define IN_K   4096
#define OUT_N  11008
#define OUT_CH 43       // 11008 / 256 row-chunks for col-norm partials
#define KSTEPS 128      // IN_K / 32

typedef __attribute__((ext_vector_type(16))) __bf16          v16bf;
typedef __attribute__((ext_vector_type(16))) unsigned short  v16u;
typedef __attribute__((ext_vector_type(8)))  float           v8f;
typedef __attribute__((ext_vector_type(4)))  int             v4i;

#define AS_GLOBAL __attribute__((address_space(1)))
#define AS_LOCAL  __attribute__((address_space(3)))

// Pack two floats to bf16 pair: round-to-nearest (+0x8000) then one v_perm_b32
// grabs both high halves. 3 VALU per pair.
__device__ __forceinline__ unsigned int pack_bf16(float lo, float hi) {
  const unsigned int ul = __builtin_bit_cast(unsigned int, lo) + 0x8000u;
  const unsigned int uh = __builtin_bit_cast(unsigned int, hi) + 0x8000u;
  return __builtin_amdgcn_perm(uh, ul, 0x07060302u);
}

// ---- CDNA5 async global->LDS copy (ASYNCcnt), with sync fallback ----------
#if __has_builtin(__builtin_amdgcn_global_load_async_to_lds_b128)
#define HAS_ASYNC_LDS 1
#else
#define HAS_ASYNC_LDS 0
#endif

__device__ __forceinline__ void async_copy16(const void* g, void* l) {
#if HAS_ASYNC_LDS
  __builtin_amdgcn_global_load_async_to_lds_b128(
      (AS_GLOBAL v4i*)g, (AS_LOCAL v4i*)l, 0, 0);
#else
  *(uint4*)l = *(const uint4*)g;
#endif
}

__device__ __forceinline__ void wait_async0() {
#if HAS_ASYNC_LDS
#if __has_builtin(__builtin_amdgcn_s_wait_asynccnt)
  __builtin_amdgcn_s_wait_asynccnt(0);
#else
  asm volatile("s_wait_asynccnt 0x0" ::: "memory");
#endif
#endif
}

// ---------------------------------------------------------------- K1a
// Partial column L1 norms: block (bx,by) reduces 256 rows for 256 cols.
__global__ __launch_bounds__(256) void colnorm_partial_kernel(
    const float* __restrict__ w, float* __restrict__ partial) {
  const int c  = blockIdx.x * 256 + threadIdx.x;
  const int o0 = blockIdx.y * 256;
  float acc = 0.0f;
  for (int o = o0; o < o0 + 256; ++o)
    acc += fabsf(w[(size_t)o * IN_K + c]);
  partial[(size_t)blockIdx.y * IN_K + c] = acc;
}

// ---------------------------------------------------------------- K1b
__global__ __launch_bounds__(256) void colnorm_reduce_kernel(
    const float* __restrict__ partial, float* __restrict__ col_norms) {
  const int c = blockIdx.x * 256 + threadIdx.x;
  float acc = 0.0f;
  for (int p = 0; p < OUT_CH; ++p)
    acc += partial[(size_t)p * IN_K + c];
  col_norms[c] = acc;
}

// ---------------------------------------------------------------- K2
// Single block: bitonic-sort the 4096 col norms in LDS, linear-interp
// quantiles (jnp.quantile 'linear'), emit modify mask + denom.
__global__ __launch_bounds__(1024) void quantile_mask_kernel(
    const float* __restrict__ col_norms, float* __restrict__ mf,
    float* __restrict__ denom_out) {
  __shared__ float s[IN_K];
  __shared__ float loHi[2];
  const int tid = threadIdx.x;

  for (int i = tid; i < IN_K; i += 1024) s[i] = col_norms[i];
  __syncthreads();

  for (int k = 2; k <= IN_K; k <<= 1) {
    for (int j = k >> 1; j > 0; j >>= 1) {
      for (int i = tid; i < IN_K; i += 1024) {
        const int l = i ^ j;
        if (l > i) {
          const bool up = ((i & k) == 0);
          const float a = s[i], b = s[l];
          if ((a > b) == up) { s[i] = b; s[l] = a; }
        }
      }
      __syncthreads();
    }
  }

  if (tid == 0) {
    // pos_lo = 0.05*4095 = 204.75 ; pos_hi = 0.95*4095 = 3890.25
    loHi[0] = s[204]  + 0.75f * (s[205]  - s[204]);
    loHi[1] = s[3890] + 0.25f * (s[3891] - s[3890]);
  }
  __syncthreads();
  const float lo = loHi[0], hi = loHi[1];

  float part = 0.0f;
  for (int i = tid; i < IN_K; i += 1024) {
    const float cn = col_norms[i];
    const float m = ((cn > lo) && (cn < hi)) ? 0.0f : 1.0f;  // modify = !outlier
    mf[i] = m;
    part += m;
  }
  __syncthreads();
  s[tid] = part;
  __syncthreads();
  for (int off = 512; off > 0; off >>= 1) {
    if (tid < off) s[tid] += s[tid + off];
    __syncthreads();
  }
  if (tid == 0) denom_out[0] = fmaxf(s[0], 1.0f);
}

// ---------------------------------------------------------------- K3
// Per-row scale: one wave32 per row, coalesced along the row.
__global__ __launch_bounds__(256) void scale_kernel(
    const float* __restrict__ w, const float* __restrict__ mf,
    const float* __restrict__ denom, float* __restrict__ scale) {
  const int lane = threadIdx.x & 31;
  const int wv   = threadIdx.x >> 5;
  const int o    = blockIdx.x * 8 + wv;
  const float* row = w + (size_t)o * IN_K;
  float acc = 0.0f;
  for (int j = lane; j < IN_K; j += 32)
    acc += fabsf(row[j]) * mf[j];
  for (int off = 16; off > 0; off >>= 1)
    acc += __shfl_xor(acc, off, 32);
  if (lane == 0) scale[o] = acc / denom[0];
}

// ---------------------------------------------------------------- K4: xpack
// Convert x once to bf16 A-fragment panels (ISA 7.12.2 layout), both raw
// (for G1) and mf-masked (for G2 = (x*mf)@w^T). Panel layout, in halfs:
//   type[0|1] -> [h][ks][tile][lane][16]   (4096 halfs = 8KB per (h,ks))
__global__ __launch_bounds__(256) void xpack_kernel(
    const float* __restrict__ x, const float* __restrict__ mf,
    unsigned short* __restrict__ xp) {
  const int g  = blockIdx.x * 256 + threadIdx.x;   // 131072 threads
  const int m  = g >> 9;                           // row 0..255
  const int c0 = (g & 511) << 3;                   // col base, multiple of 8
  const float4 xa = *(const float4*)(x + (size_t)m * IN_K + c0);
  const float4 xb = *(const float4*)(x + (size_t)m * IN_K + c0 + 4);
  const float4 ma = *(const float4*)(mf + c0);
  const float4 mb = *(const float4*)(mf + c0 + 4);

  const int h    = m >> 7;
  const int t    = (m >> 4) & 7;
  const int ks   = c0 >> 5;
  const int lane = (m & 15) | (((c0 >> 3) & 1) << 4);
  const int elp  = (c0 >> 4) & 1;  // element-pair base 0 or 4 (uints)

  unsigned int* dst = (unsigned int*)xp
      + (size_t)((h * KSTEPS + ks) * 2048) + (((t << 5) | lane) << 3) + (elp << 2);

  uint4 p1, p2;
  p1.x = pack_bf16(xa.x, xa.y);
  p1.y = pack_bf16(xa.z, xa.w);
  p1.z = pack_bf16(xb.x, xb.y);
  p1.w = pack_bf16(xb.z, xb.w);
  *(uint4*)dst = p1;
  p2.x = pack_bf16(xa.x * ma.x, xa.y * ma.y);
  p2.y = pack_bf16(xa.z * ma.z, xa.w * ma.w);
  p2.z = pack_bf16(xb.x * mb.x, xb.y * mb.y);
  p2.w = pack_bf16(xb.z * mb.z, xb.w * mb.w);
  *(uint4*)(dst + 524288) = p2;   // second panel-type region (2MB / 4B)
}

// ---------------------------------------------------------------- K5: GEMM
// out[m,n] = G1[m,n] + (scale[n]-1)*G2[m,n] + bias[n]
// grid (2, 172), 128 threads (4 waves). Wave: one 16-wide N tile, 8 M-tiles,
// two accumulator sets (G1, G2). Panels double-buffered in LDS via
// GLOBAL_LOAD_ASYNC_TO_LDS_B128; next-step w float4s preloaded to registers;
// A fragments software-pipelined by one tile so ds_load latency hides under
// the previous tile's XDL WMMAs.
__global__ __launch_bounds__(128) void gemm_wmma_kernel(
    const unsigned short* __restrict__ xp, const float* __restrict__ w,
    const float* __restrict__ bias, const float* __restrict__ scale,
    float* __restrict__ out) {
  __shared__ v16u aLds[2 * 2 * 256];   // [buf][panel][tile*32+lane] = 32KB

  const int tid   = threadIdx.x;
  const int lane  = tid & 31;
  const int wv    = tid >> 5;
  const int h     = blockIdx.x;         // M half
  const int mBase = h * 128;
  const int n     = blockIdx.y * 64 + wv * 16 + (lane & 15);
  const int c0    = (lane < 16) ? 0 : 8;

  const float s   = scale[n];
  const float sm1 = s - 1.0f;
  const float* wrowBase = w + (size_t)n * IN_K + c0;

  const unsigned short* xp1 = xp + (size_t)h * (KSTEPS * 4096);
  const unsigned short* xp2 = xp1 + (size_t)2 * KSTEPS * 4096;
  char* ldsByte = (char*)aLds;

  v8f acc1[8], acc2[8];
  #pragma unroll
  for (int t = 0; t < 8; ++t) { v8f z = {}; acc1[t] = z; acc2[t] = z; }

  // prologue: stage panel 0 into buffer 0; preload B for step 0
  {
    const char* s1 = (const char*)xp1 + tid * 64;
    const char* s2 = (const char*)xp2 + tid * 64;
    char* d1 = ldsByte + tid * 64;
    char* d2 = ldsByte + 8192 + tid * 64;
    async_copy16(s1, d1);           async_copy16(s1 + 16, d1 + 16);
    async_copy16(s1 + 32, d1 + 32); async_copy16(s1 + 48, d1 + 48);
    async_copy16(s2, d2);           async_copy16(s2 + 16, d2 + 16);
    async_copy16(s2 + 32, d2 + 32); async_copy16(s2 + 48, d2 + 48);
  }
  float4 nb0 = *(const float4*)(wrowBase);
  float4 nb1 = *(const float4*)(wrowBase + 4);
  float4 nb2 = *(const float4*)(wrowBase + 16);
  float4 nb3 = *(const float4*)(wrowBase + 20);

  for (int ks = 0; ks < KSTEPS; ++ks) {
    wait_async0();        // my panel-ks asyncs complete
    __syncthreads();      // everyone's complete; prior reads of other buf done
    const int cur = ks & 1;

    if (ks + 1 < KSTEPS) {  // stage panel ks+1 into the other buffer
      const char* s1 = (const char*)(xp1 + (ks + 1) * 4096) + tid * 64;
      const char* s2 = (const char*)(xp2 + (ks + 1) * 4096) + tid * 64;
      char* d1 = ldsByte + (cur ^ 1) * 16384 + tid * 64;
      char* d2 = d1 + 8192;
      async_copy16(s1, d1);           async_copy16(s1 + 16, d1 + 16);
      async_copy16(s1 + 32, d1 + 32); async_copy16(s1 + 48, d1 + 48);
      async_copy16(s2, d2);           async_copy16(s2 + 16, d2 + 16);
      async_copy16(s2 + 32, d2 + 32); async_copy16(s2 + 48, d2 + 48);
    }

    const float4 b0 = nb0, b1 = nb1, b2 = nb2, b3 = nb3;
    if (ks + 1 < KSTEPS) {  // preload next-step w fragment (hides latency)
      const float* wr = wrowBase + (ks + 1) * 32;
      nb0 = *(const float4*)(wr);
      nb1 = *(const float4*)(wr + 4);
      nb2 = *(const float4*)(wr + 16);
      nb3 = *(const float4*)(wr + 20);
    }

    // raw-w B fragment: 16 adds + 8 perms, no mask/scale work
    v16u bu;
    unsigned int* bup = (unsigned int*)&bu;
    bup[0] = pack_bf16(b0.x, b0.y); bup[1] = pack_bf16(b0.z, b0.w);
    bup[2] = pack_bf16(b1.x, b1.y); bup[3] = pack_bf16(b1.z, b1.w);
    bup[4] = pack_bf16(b2.x, b2.y); bup[5] = pack_bf16(b2.z, b2.w);
    bup[6] = pack_bf16(b3.x, b3.y); bup[7] = pack_bf16(b3.z, b3.w);
    const v16bf bfr = __builtin_bit_cast(v16bf, bu);

    // A fragments: pipelined by one tile (independent regs -> ds loads for
    // tile t+1 issue before tile t's WMMAs, partial dscnt waits suffice).
    const v16u* bc = aLds + cur * 512;
    v16u pa1 = bc[lane];
    v16u pa2 = bc[256 | lane];
    #pragma unroll
    for (int t = 0; t < 8; ++t) {
      const v16u ca1 = pa1;
      const v16u ca2 = pa2;
      if (t < 7) {
        pa1 = bc[((t + 1) << 5) | lane];
        pa2 = bc[256 | (((t + 1) << 5) | lane)];
      }
      acc1[t] = __builtin_amdgcn_wmma_f32_16x16x32_bf16(
          false, __builtin_bit_cast(v16bf, ca1), false, bfr, (short)0,
          acc1[t], false, false);
      acc2[t] = __builtin_amdgcn_wmma_f32_16x16x32_bf16(
          false, __builtin_bit_cast(v16bf, ca2), false, bfr, (short)0,
          acc2[t], false, false);
    }
  }

  const float bb = bias[n];
  #pragma unroll
  for (int t = 0; t < 8; ++t) {
    #pragma unroll
    for (int g = 0; g < 8; ++g) {
      const int m = mBase + (t << 4) + g + ((lane >> 4) << 3);
      out[(size_t)m * OUT_N + n] = fmaf(sm1, acc2[t][g], acc1[t][g]) + bb;
    }
  }
}

// ---------------------------------------------------------------- launch
extern "C" void kernel_launch(void* const* d_in, const int* in_sizes, int n_in,
                              void* d_out, int out_size, void* d_ws, size_t ws_size,
                              hipStream_t stream) {
  (void)in_sizes; (void)n_in; (void)out_size; (void)ws_size;
  const float* x    = (const float*)d_in[0];
  const float* w    = (const float*)d_in[1];
  const float* bias = (const float*)d_in[2];
  float* out = (float*)d_out;

  float* ws        = (float*)d_ws;
  float* partial   = ws;                               // 43*4096
  float* col_norms = partial + (size_t)OUT_CH * IN_K;  // 4096
  float* mf        = col_norms + IN_K;                 // 4096
  float* denom     = mf + IN_K;                        // 1 (+pad)
  float* scale     = denom + 32;                       // 11008
  unsigned short* xp = (unsigned short*)(scale + OUT_N);  // 4MB bf16 panels

  colnorm_partial_kernel<<<dim3(IN_K / 256, OUT_CH), 256, 0, stream>>>(w, partial);
  colnorm_reduce_kernel<<<IN_K / 256, 256, 0, stream>>>(partial, col_norms);
  quantile_mask_kernel<<<1, 1024, 0, stream>>>(col_norms, mf, denom);
  scale_kernel<<<OUT_N / 8, 256, 0, stream>>>(w, mf, denom, scale);
  xpack_kernel<<<512, 256, 0, stream>>>(x, mf, xp);
  gemm_wmma_kernel<<<dim3(2, OUT_N / 64), 128, 0, stream>>>(xp, w, bias, scale, out);
}
